// TransformerDecoder_83983790506082
// MI455X (gfx1250) — matrix operand order repaired and checked by reading, not compiled, not run on hardware
//
#include <hip/hip_runtime.h>
#include <hip/hip_bf16.h>
#include <math.h>

// ---------------------------------------------------------------- constants
#define D_MODEL 1024
#define NHEAD   16
#define DHEAD   64
#define BATCH   4
#define SEQ     1024
#define NSLOTS  32

typedef __bf16 v16bf __attribute__((ext_vector_type(16)));
typedef __bf16 v8bf  __attribute__((ext_vector_type(8)));
typedef float  v8f   __attribute__((ext_vector_type(8)));

// ---------------------------------------------------------------- helpers
__device__ __forceinline__ unsigned short f2bf_bits(float f) {
    __bf16 h = (__bf16)f;                       // native cvt (RNE)
    return __builtin_bit_cast(unsigned short, h);
}
__device__ __forceinline__ __bf16 bits2bf(unsigned short s) {
    return __builtin_bit_cast(__bf16, s);
}

__device__ __forceinline__ v8f wmma_bf16(v16bf a, v16bf b, v8f c) {
    return __builtin_amdgcn_wmma_f32_16x16x32_bf16(
        /*neg_a=*/false, a, /*neg_b=*/false, b,
        /*c_mod=*/(short)0, c, /*reuse_a=*/false, /*reuse_b=*/false);
}

// A-fragment (16x32 bf16) from row-major bf16, p = row + k0 + kg*8.
// elems 0..7 -> K = kg*8+i (p[0..7]); elems 8..15 -> K = 16+kg*8+i (p[16..23])
__device__ __forceinline__ v16bf load_a_frag_bf(const unsigned short* __restrict__ p) {
    v8bf lo = *(const v8bf*)(p);
    v8bf hi = *(const v8bf*)(p + 16);
    return __builtin_shufflevector(lo, hi, 0, 1, 2, 3, 4, 5, 6, 7,
                                           8, 9, 10, 11, 12, 13, 14, 15);
}
// B-fragment (32x16), p = col + k0 + kg*8: elems 0..7 -> p[0..7], elems 8..15 -> p[16..23]
__device__ __forceinline__ v16bf load_b_frag_bf(const unsigned short* __restrict__ p) {
    v8bf lo = *(const v8bf*)(p);
    v8bf hi = *(const v8bf*)(p + 16);
    return __builtin_shufflevector(lo, hi, 0, 1, 2, 3, 4, 5, 6, 7,
                                           8, 9, 10, 11, 12, 13, 14, 15);
}

// ---------------------------------------------------------------- kernels
__global__ void add_pos_kernel(const float* __restrict__ inp,
                               const float* __restrict__ pos,
                               float* __restrict__ x,
                               unsigned short* __restrict__ xbf, int total) {
    int i = blockIdx.x * blockDim.x + threadIdx.x;
    if (i < total) {
        float v = inp[i] + pos[i & (SEQ * D_MODEL - 1)];
        x[i] = v;
        xbf[i] = f2bf_bits(v);
    }
}

__global__ void conv_kernel(const float* __restrict__ a,
                            unsigned short* __restrict__ o, int total) {
    int i = blockIdx.x * blockDim.x + threadIdx.x;
    if (i < total) o[i] = f2bf_bits(a[i]);
}

// W f32 [K=1024, N=1024] row-major  ->  WT bf16 [N, K] row-major
__global__ void convt_kernel(const float* __restrict__ W,
                             unsigned short* __restrict__ WT) {
    int i = blockIdx.x * blockDim.x + threadIdx.x;   // 0 .. 1M-1
    int n  = i >> 10;
    int kk = i & 1023;
    WT[i] = f2bf_bits(W[(kk << 10) | n]);
}

// epilogue(A_bf[M,1024] @ WT^T + bias):
//   MODE 1: val = acc + bias[n]
//   MODE 2: val = res[m,n] + relu(acc + bias[n])
//   HAS_C -> f32 store, HAS_CBF -> bf16*cbfscale store  (compile-time variants)
template <int MODE, bool HAS_C, bool HAS_CBF>
__global__ __launch_bounds__(128) void gemm_bf16_kernel(
    const unsigned short* __restrict__ Abf, const unsigned short* __restrict__ WT,
    const float* __restrict__ bias, const float* __restrict__ res,
    float* __restrict__ C, unsigned short* __restrict__ Cbf, float cbfscale) {
    const int K = 1024, N = 1024;
    int lane = threadIdx.x & 31;
    int wave = threadIdx.x >> 5;
    int m_base = blockIdx.y * 64 + (wave >> 1) * 32;     // wave tile 32 x 64
    int n_base = blockIdx.x * 128 + (wave & 1) * 64;
    int rowl = lane & 15;
    int kg   = lane >> 4;

    v8f acc[2][4];
#pragma unroll
    for (int ti = 0; ti < 2; ++ti)
#pragma unroll
        for (int tj = 0; tj < 4; ++tj)
            acc[ti][tj] = (v8f){0.f, 0.f, 0.f, 0.f, 0.f, 0.f, 0.f, 0.f};

    for (int k0 = 0; k0 < K; k0 += 32) {
        v16bf a[2], b[4];
#pragma unroll
        for (int ti = 0; ti < 2; ++ti)
            a[ti] = load_a_frag_bf(
                Abf + (size_t)(m_base + ti * 16 + rowl) * K + k0 + kg * 8);
#pragma unroll
        for (int tj = 0; tj < 4; ++tj)
            b[tj] = load_b_frag_bf(
                WT + (size_t)(n_base + tj * 16 + rowl) * K + k0 + kg * 8);
#pragma unroll
        for (int ti = 0; ti < 2; ++ti)
#pragma unroll
            for (int tj = 0; tj < 4; ++tj)
                acc[ti][tj] = wmma_bf16(a[ti], b[tj], acc[ti][tj]);
    }

#pragma unroll
    for (int ti = 0; ti < 2; ++ti)
#pragma unroll
        for (int tj = 0; tj < 4; ++tj) {
            int n = n_base + tj * 16 + rowl;
            float bv = bias[n];
            size_t base = (size_t)(m_base + ti * 16 + 8 * kg) * N + n;
#pragma unroll
            for (int r = 0; r < 8; ++r) {        // C layout row = r + 8*(lane/16)
                size_t idx = base + (size_t)r * N;
                float val = acc[ti][tj][r] + bv;
                if constexpr (MODE == 2) val = res[idx] + fmaxf(val, 0.f);
                if constexpr (HAS_C)   C[idx] = val;
                if constexpr (HAS_CBF) Cbf[idx] = f2bf_bits(val * cbfscale);
            }
        }
}

// Flash-style attention, all-bf16 operands: 1 wave per (b, h, 16-query tile).
// qbf already carries the 1/sqrt(dh) scale. SK is a compile-time key count.
template <int SK>
__global__ __launch_bounds__(32) void attn_kernel(
    const unsigned short* __restrict__ qbf, const unsigned short* __restrict__ kbf,
    const unsigned short* __restrict__ vbf, float* __restrict__ out) {
    const int D = D_MODEL;
    int lane = threadIdx.x & 31;
    int rowl = lane & 15;
    int kg   = lane >> 4;
    int q0 = blockIdx.x * 16;
    int h  = blockIdx.y;
    int b  = blockIdx.z;

    const unsigned short* qbase = qbf + ((size_t)(b * SEQ + q0)) * D + h * DHEAD;
    const unsigned short* kbase = kbf + (size_t)b * SK * D + h * DHEAD;
    const unsigned short* vbase = vbf + (size_t)b * SK * D + h * DHEAD;
    float*                obase = out + ((size_t)(b * SEQ + q0)) * D + h * DHEAD;

    v16bf aq[2];
#pragma unroll
    for (int c = 0; c < 2; ++c)
        aq[c] = load_a_frag_bf(qbase + rowl * D + c * 32 + kg * 8);

    v8f o[4];
#pragma unroll
    for (int dc = 0; dc < 4; ++dc)
        o[dc] = (v8f){0.f, 0.f, 0.f, 0.f, 0.f, 0.f, 0.f, 0.f};
    float mrow[8], lrow[8];
#pragma unroll
    for (int r = 0; r < 8; ++r) { mrow[r] = -1e30f; lrow[r] = 0.f; }

    __shared__ unsigned short pbuf[16 * 32];     // P tile [query row][local key]

    for (int kb = 0; kb < SK; kb += 32) {
        // ---- scores: two 16x16 C-tiles over 32 keys, dh summed in 2 chunks
        v8f s[2];
#pragma unroll
        for (int kh = 0; kh < 2; ++kh) {
            v8f sc = (v8f){0.f, 0.f, 0.f, 0.f, 0.f, 0.f, 0.f, 0.f};
#pragma unroll
            for (int kc = 0; kc < 2; ++kc) {
                // K^T fragment: lane = key column, elems = dh in table order
                v16bf bk = load_b_frag_bf(kbase +
                    (size_t)(kb + kh * 16 + rowl) * D + kc * 32 + kg * 8);
                sc = wmma_bf16(aq[kc], bk, sc);
            }
            s[kh] = sc;
        }
        // ---- online softmax (rows = r + 8*kg; 16-lane row groups)
#pragma unroll
        for (int r = 0; r < 8; ++r) {
            float mx = fmaxf(s[0][r], s[1][r]);
#pragma unroll
            for (int off = 8; off > 0; off >>= 1)
                mx = fmaxf(mx, __shfl_xor(mx, off, 32));
            float mnew  = fmaxf(mrow[r], mx);
            float alpha = __expf(mrow[r] - mnew);
            float p0 = __expf(s[0][r] - mnew);
            float p1 = __expf(s[1][r] - mnew);
            float rs = p0 + p1;
#pragma unroll
            for (int off = 8; off > 0; off >>= 1)
                rs += __shfl_xor(rs, off, 32);
            lrow[r] = lrow[r] * alpha + rs;
            mrow[r] = mnew;
            s[0][r] = p0; s[1][r] = p1;
#pragma unroll
            for (int dc = 0; dc < 4; ++dc)
                o[dc][r] = o[dc][r] * alpha;
        }
        // ---- C-layout P -> LDS -> A-fragment layout
#pragma unroll
        for (int r = 0; r < 8; ++r) {
            int prow = r + 8 * kg;
            pbuf[prow * 32 + rowl]      = f2bf_bits(s[0][r]);
            pbuf[prow * 32 + 16 + rowl] = f2bf_bits(s[1][r]);
        }
        __syncthreads();
        v16bf pa;
#pragma unroll
        for (int i = 0; i < 8; ++i) {
            pa[i]     = bits2bf(pbuf[rowl * 32 + kg * 8 + i]);
            pa[8 + i] = bits2bf(pbuf[rowl * 32 + 16 + kg * 8 + i]);
        }
        __syncthreads();
        // ---- O += P @ V   (4 dh column tiles)
#pragma unroll
        for (int dc = 0; dc < 4; ++dc) {
            v16bf bv;
#pragma unroll
            for (int i = 0; i < 16; ++i) {
                int key = kb + (i < 8 ? kg * 8 + i : 16 + kg * 8 + (i - 8));
                bv[i] = bits2bf(vbase[(size_t)key * D + dc * 16 + rowl]);
            }
            o[dc] = wmma_bf16(pa, bv, o[dc]);
        }
    }
    // ---- normalize and write
#pragma unroll
    for (int r = 0; r < 8; ++r) {
        float inv = 1.0f / lrow[r];
        int row = r + 8 * kg;
#pragma unroll
        for (int dc = 0; dc < 4; ++dc)
            obase[(size_t)row * D + dc * 16 + rowl] = o[dc][r] * inv;
    }
}

// out = LN(a (+ badd)) * g + beta ; optional bf16 mirror   (D=1024, block=256)
template <bool HAS_ADD, bool HAS_BF>
__global__ __launch_bounds__(256) void ln_kernel(
    const float* __restrict__ a, const float* __restrict__ badd,
    const float* __restrict__ g, const float* __restrict__ beta,
    float* __restrict__ out, unsigned short* __restrict__ outbf) {
    int row = blockIdx.x;
    int t = threadIdx.x;
    float4 x = *(const float4*)(a + (size_t)row * D_MODEL + t * 4);
    if constexpr (HAS_ADD) {
        float4 y = *(const float4*)(badd + (size_t)row * D_MODEL + t * 4);
        x.x += y.x; x.y += y.y; x.z += y.z; x.w += y.w;
    }
    __shared__ float red[256];
    red[t] = x.x + x.y + x.z + x.w;
    __syncthreads();
    for (int off = 128; off > 0; off >>= 1) {
        if (t < off) red[t] += red[t + off];
        __syncthreads();
    }
    float mean = red[0] * (1.0f / 1024.0f);
    __syncthreads();
    float dx = x.x - mean, dy = x.y - mean, dz = x.z - mean, dw = x.w - mean;
    red[t] = dx * dx + dy * dy + dz * dz + dw * dw;
    __syncthreads();
    for (int off = 128; off > 0; off >>= 1) {
        if (t < off) red[t] += red[t + off];
        __syncthreads();
    }
    float rstd = rsqrtf(red[0] * (1.0f / 1024.0f) + 1e-5f);
    float4 gg = *(const float4*)(g + t * 4);
    float4 bb = *(const float4*)(beta + t * 4);
    float4 o;
    o.x = dx * rstd * gg.x + bb.x;
    o.y = dy * rstd * gg.y + bb.y;
    o.z = dz * rstd * gg.z + bb.z;
    o.w = dw * rstd * gg.w + bb.w;
    *(float4*)(out + (size_t)row * D_MODEL + t * 4) = o;
    if constexpr (HAS_BF) {
        unsigned short* p = outbf + (size_t)row * D_MODEL + t * 4;
        p[0] = f2bf_bits(o.x); p[1] = f2bf_bits(o.y);
        p[2] = f2bf_bits(o.z); p[3] = f2bf_bits(o.w);
    }
}

// ---------------------------------------------------------------- launcher
extern "C" void kernel_launch(void* const* d_in, const int* in_sizes, int n_in,
                              void* d_out, int out_size, void* d_ws, size_t ws_size,
                              hipStream_t stream) {
    (void)in_sizes; (void)n_in; (void)out_size; (void)ws_size;
    const int M = BATCH * SEQ;                 // 4096 rows
    const size_t MD = (size_t)M * D_MODEL;     // 4M elements

    float* x    = (float*)d_ws;                // f32 activations
    float* q    = x + MD;                      // f32 q (residual)
    float* tbuf = q + MD;                      // f32 scratch (attn out / mlp)
    unsigned short* xbf  = (unsigned short*)(tbuf + MD);
    unsigned short* qbf  = xbf + MD;           // q * 1/sqrt(dh), bf16
    unsigned short* kbf  = qbf + MD;
    unsigned short* vbf  = kbf + MD;
    unsigned short* sbf  = vbf + MD;           // slots bf16 (128K)
    unsigned short* wbf  = sbf + (size_t)BATCH * NSLOTS * D_MODEL;  // weight slot

    const float* slots = (const float*)d_in[0];
    const float* inseq = (const float*)d_in[1];
    const float* pos   = (const float*)d_in[2];

    add_pos_kernel<<<(int)((MD + 255) / 256), 256, 0, stream>>>(
        inseq, pos, x, xbf, (int)MD);
    conv_kernel<<<(BATCH * NSLOTS * D_MODEL + 255) / 256, 256, 0, stream>>>(
        slots, sbf, BATCH * NSLOTS * D_MODEL);

    for (int layer = 0; layer < 4; ++layer) {
        for (int pass = 0; pass < 2; ++pass) {   // 0 = self, 1 = cross(slots)
            int base = 3 + layer * 24 + pass * 12;
            const float* Wq = (const float*)d_in[base + 0];
            const float* bq = (const float*)d_in[base + 1];
            const float* Wk = (const float*)d_in[base + 2];
            const float* bk = (const float*)d_in[base + 3];
            const float* Wv = (const float*)d_in[base + 4];
            const float* bv = (const float*)d_in[base + 5];
            const float* Wo = (const float*)d_in[base + 6];
            const float* bo = (const float*)d_in[base + 7];
            const float* g0 = (const float*)d_in[base + 8];
            const float* b0 = (const float*)d_in[base + 9];
            const float* g1 = (const float*)d_in[base + 10];
            const float* b1 = (const float*)d_in[base + 11];

            const unsigned short* kin = (pass == 0) ? xbf : sbf;
            int Mk = (pass == 0) ? M : BATCH * NSLOTS;     // 4096 or 128

            // q = x@Wq + bq  (f32 for residual; bf16 mirror pre-scaled by 1/8)
            convt_kernel<<<4096, 256, 0, stream>>>(Wq, wbf);
            gemm_bf16_kernel<1, true, true><<<dim3(8, M / 64), 128, 0, stream>>>(
                xbf, wbf, bq, nullptr, q, qbf, 0.125f);
            // k, v: bf16 only (consumed solely by attention)
            convt_kernel<<<4096, 256, 0, stream>>>(Wk, wbf);
            gemm_bf16_kernel<1, false, true><<<dim3(8, Mk / 64), 128, 0, stream>>>(
                kin, wbf, bk, nullptr, nullptr, kbf, 1.0f);
            convt_kernel<<<4096, 256, 0, stream>>>(Wv, wbf);
            gemm_bf16_kernel<1, false, true><<<dim3(8, Mk / 64), 128, 0, stream>>>(
                kin, wbf, bv, nullptr, nullptr, vbf, 1.0f);

            if (pass == 0)
                attn_kernel<SEQ><<<dim3(SEQ / 16, NHEAD, BATCH), 32, 0, stream>>>(
                    qbf, kbf, vbf, tbuf);
            else
                attn_kernel<NSLOTS><<<dim3(SEQ / 16, NHEAD, BATCH), 32, 0, stream>>>(
                    qbf, kbf, vbf, tbuf);

            // x = LN(q + attn)
            ln_kernel<true, true><<<M, 256, 0, stream>>>(q, tbuf, g0, b0, x, xbf);

            // t = x + relu(x@Wo + bo)
            convt_kernel<<<4096, 256, 0, stream>>>(Wo, wbf);
            gemm_bf16_kernel<2, true, false><<<dim3(8, M / 64), 128, 0, stream>>>(
                xbf, wbf, bo, x, tbuf, nullptr, 1.0f);

            // x = LN(t)
            ln_kernel<false, true><<<M, 256, 0, stream>>>(tbuf, nullptr, g1, b1, x, xbf);
        }
    }

    convt_kernel<<<4096, 256, 0, stream>>>((const float*)d_in[99], wbf);
    gemm_bf16_kernel<1, true, false><<<dim3(8, M / 64), 128, 0, stream>>>(
        xbf, wbf, (const float*)d_in[100], nullptr, (float*)d_out, nullptr, 1.0f);
}
